// SCM_module_48868137894390
// MI455X (gfx1250) — compile-verified, matrix operand order checked by reading, not run on hardware
//
#include <hip/hip_runtime.h>
#include <hip/hip_bf16.h>
#include <math.h>

typedef __attribute__((ext_vector_type(16))) __bf16 v16bf;
typedef __attribute__((ext_vector_type(8)))  float  v8f;

// Fragment loaders. p points at row_start + k0 + 8*half; elements j=0..7 come
// from p[0..7], j=8..15 from p[16..23] (ISA 16-bit 16x32 operand layout).

// Fast path: direct f32->bf16 convert (tile fully in-bounds).
__device__ __forceinline__ v16bf load_frag16(const float* __restrict__ p) {
    float4 a0 = ((const float4*)p)[0];
    float4 a1 = ((const float4*)p)[1];
    float4 b0 = ((const float4*)(p + 16))[0];
    float4 b1 = ((const float4*)(p + 16))[1];
    v16bf f;
    f[0]  = (__bf16)a0.x; f[1]  = (__bf16)a0.y;
    f[2]  = (__bf16)a0.z; f[3]  = (__bf16)a0.w;
    f[4]  = (__bf16)a1.x; f[5]  = (__bf16)a1.y;
    f[6]  = (__bf16)a1.z; f[7]  = (__bf16)a1.w;
    f[8]  = (__bf16)b0.x; f[9]  = (__bf16)b0.y;
    f[10] = (__bf16)b0.z; f[11] = (__bf16)b0.w;
    f[12] = (__bf16)b1.x; f[13] = (__bf16)b1.y;
    f[14] = (__bf16)b1.z; f[15] = (__bf16)b1.w;
    return f;
}

// Edge path: s = 1.0 for valid rows, 0.0 for clamped rows -> zeros, branchless.
__device__ __forceinline__ v16bf load_frag16_s(const float* __restrict__ p, float s) {
    float4 a0 = ((const float4*)p)[0];
    float4 a1 = ((const float4*)p)[1];
    float4 b0 = ((const float4*)(p + 16))[0];
    float4 b1 = ((const float4*)(p + 16))[1];
    v16bf f;
    f[0]  = (__bf16)(a0.x * s); f[1]  = (__bf16)(a0.y * s);
    f[2]  = (__bf16)(a0.z * s); f[3]  = (__bf16)(a0.w * s);
    f[4]  = (__bf16)(a1.x * s); f[5]  = (__bf16)(a1.y * s);
    f[6]  = (__bf16)(a1.z * s); f[7]  = (__bf16)(a1.w * s);
    f[8]  = (__bf16)(b0.x * s); f[9]  = (__bf16)(b0.y * s);
    f[10] = (__bf16)(b0.z * s); f[11] = (__bf16)(b0.w * s);
    f[12] = (__bf16)(b1.x * s); f[13] = (__bf16)(b1.y * s);
    f[14] = (__bf16)(b1.z * s); f[15] = (__bf16)(b1.w * s);
    return f;
}

// ---------------------------------------------------------------------------
// Generic WMMA GEMM:  C[M,N] = A[M,K] @ W[N,K]^T   (fp32 in, bf16 WMMA, fp32 acc)
// Requirements: K % 32 == 0; lda/ldw multiples of 4 floats (16B rows).
// Each wave computes a 64x16 output strip (4 M-tiles x 1 N-tile); the weight
// fragment is loaded once per K-step and reused by 4 WMMAs.
// Tiles fully in-bounds take a scale-free fast path; edge tiles (N not a
// multiple of 16, M tail) use row clamping + zero-scaling. Both paths are
// wave-uniform, so EXEC stays all-ones for every WMMA.
// ---------------------------------------------------------------------------
__global__ void wmma_gemm_atb(const float* __restrict__ A, int lda,
                              const float* __restrict__ W, int ldw,
                              float* __restrict__ C, int ldc,
                              int M, int N, int K, int tilesN, int tilesM4) {
    int wave = (int)((blockIdx.x * blockDim.x + threadIdx.x) >> 5);
    int lane = threadIdx.x & 31;
    if (wave >= tilesN * tilesM4) return;   // whole-wave exit: EXEC uniform

    const int tn      = wave % tilesN;
    const int rowBase = (wave / tilesN) * 64;
    const int col0    = tn * 16;
    const int ln16    = lane & 15;
    const int half8   = (lane >> 4) << 3;   // 0 or 8

    const int nW = col0 + ln16;
    const float wscale = (nW < N) ? 1.f : 0.f;
    const float* __restrict__ Wrow =
        W + (size_t)((nW < N) ? nW : (N - 1)) * ldw + half8;

    const float* __restrict__ Arow[4];
    float ascale[4];
#pragma unroll
    for (int t = 0; t < 4; ++t) {
        int m = rowBase + t * 16 + ln16;
        ascale[t] = (m < M) ? 1.f : 0.f;
        Arow[t] = A + (size_t)((m < M) ? m : (M - 1)) * lda + half8;
    }

    v8f acc[4] = {};
    if (col0 + 16 <= N && rowBase + 64 <= M) {
        // ---- fast path: whole 64x16 strip in-bounds, no scaling ----
        for (int k0 = 0; k0 < K; k0 += 32) {
            __builtin_prefetch(Wrow + k0 + 32, 0, 1);      // global_prefetch_b8
            __builtin_prefetch(Arow[0] + k0 + 32, 0, 1);
            v16bf bfrag = load_frag16(Wrow + k0);
#pragma unroll
            for (int t = 0; t < 4; ++t) {
                v16bf afrag = load_frag16(Arow[t] + k0);
                acc[t] = __builtin_amdgcn_wmma_f32_16x16x32_bf16(
                    false, afrag, false, bfrag, (short)0, acc[t], false, false);
            }
        }
    } else {
        // ---- edge path: clamped rows contribute zeros via scale ----
        for (int k0 = 0; k0 < K; k0 += 32) {
            v16bf bfrag = load_frag16_s(Wrow + k0, wscale);
#pragma unroll
            for (int t = 0; t < 4; ++t) {
                v16bf afrag = load_frag16_s(Arow[t] + k0, ascale[t]);
                acc[t] = __builtin_amdgcn_wmma_f32_16x16x32_bf16(
                    false, afrag, false, bfrag, (short)0, acc[t], false, false);
            }
        }
    }

#pragma unroll
    for (int t = 0; t < 4; ++t) {
#pragma unroll
        for (int r = 0; r < 8; ++r) {
            int m = rowBase + t * 16 + r + half8;
            if (m < M && nW < N) C[(size_t)m * ldc + nW] = acc[t][r];
        }
    }
}

// ---------------------------------------------------------------------------
// LayerNorm over channel dim with transpose: x[B,C,L] -> y[B,L,C], norm over C.
// One 128-thread block per (b,l); C == 128.
// ---------------------------------------------------------------------------
__global__ void ln_over_c(const float* __restrict__ x, const float* __restrict__ g,
                          const float* __restrict__ bb, float* __restrict__ y,
                          int Cn, int Ln) {
    int bl = blockIdx.x;                 // b*Ln + l
    int b = bl / Ln, l = bl % Ln;
    int c = threadIdx.x;                 // blockDim.x == Cn == 128
    float v = x[((size_t)b * Cn + c) * Ln + l];

    __shared__ float sm[128];
    sm[c] = v; __syncthreads();
    for (int s = 64; s > 0; s >>= 1) { if (c < s) sm[c] += sm[c + s]; __syncthreads(); }
    float mu = sm[0] / (float)Cn;
    __syncthreads();
    float d = v - mu;
    sm[c] = d * d; __syncthreads();
    for (int s = 64; s > 0; s >>= 1) { if (c < s) sm[c] += sm[c + s]; __syncthreads(); }
    float var = sm[0] / (float)Cn;
    float r = rsqrtf(var + 1e-5f);
    y[((size_t)b * Ln + l) * Cn + c] = d * r * g[c] + bb[c];
}

// ---------------------------------------------------------------------------
// LayerNorm over L: x[B,C,L] -> y[B,C,L], norm over last dim (len Ln).
// One 256-thread block per (b,c) row.
// ---------------------------------------------------------------------------
__global__ void ln_over_l(const float* __restrict__ x, const float* __restrict__ g,
                          const float* __restrict__ bb, float* __restrict__ y, int Ln) {
    int row = blockIdx.x;
    const float* xr = x + (size_t)row * Ln;
    float*       yr = y + (size_t)row * Ln;
    float s = 0.f, s2 = 0.f;
    for (int i = threadIdx.x; i < Ln; i += blockDim.x) {
        float v = xr[i]; s += v; s2 += v * v;
    }
    __shared__ float sa[256], sb[256];
    sa[threadIdx.x] = s; sb[threadIdx.x] = s2; __syncthreads();
    for (int st = 128; st > 0; st >>= 1) {
        if ((int)threadIdx.x < st) {
            sa[threadIdx.x] += sa[threadIdx.x + st];
            sb[threadIdx.x] += sb[threadIdx.x + st];
        }
        __syncthreads();
    }
    float mu = sa[0] / (float)Ln;
    float var = sb[0] / (float)Ln - mu * mu;
    float r = rsqrtf(var + 1e-5f);
    for (int i = threadIdx.x; i < Ln; i += blockDim.x)
        yr[i] = (xr[i] - mu) * r * g[i] + bb[i];
}

// ---------------------------------------------------------------------------
// Depthwise causal conv (K=4) along L + SiLU.
// xz[B,L,ldx] (xi = columns 0..D-1), out[B,L,D].
// ---------------------------------------------------------------------------
__global__ void conv_silu(const float* __restrict__ xz, int ldx,
                          const float* __restrict__ w, const float* __restrict__ bias,
                          float* __restrict__ out, int Bn, int Ln, int Dn) {
    int idx = blockIdx.x * blockDim.x + threadIdx.x;
    int total = Bn * Ln * Dn;
    if (idx >= total) return;
    int d = idx % Dn;
    int l = (idx / Dn) % Ln;
    int b = idx / (Dn * Ln);
    float acc = bias[d];
#pragma unroll
    for (int t = 0; t < 4; ++t) {
        int ls = l - 3 + t;
        if (ls >= 0) acc += w[d * 4 + t] * xz[((size_t)b * Ln + ls) * ldx + d];
    }
    out[idx] = acc / (1.f + expf(-acc));   // silu
}

// ---------------------------------------------------------------------------
// In-place delta = softplus(v + dt_b[d])   (used for m2, after WMMA GEMM)
// ---------------------------------------------------------------------------
__global__ void bias_softplus(float* __restrict__ v, const float* __restrict__ bias,
                              int total, int Dn) {
    int i = blockIdx.x * blockDim.x + threadIdx.x;
    if (i >= total) return;
    float x = v[i] + bias[i % Dn];
    v[i] = (x > 20.f) ? x : log1pf(expf(x));
}

// ---------------------------------------------------------------------------
// Small-rank delta projection (m1: dt_rank = 8, too small for WMMA):
// delta[row,d] = softplus(sum_r dbl[row,r] * dt_w[d,r] + dt_b[d])
// ---------------------------------------------------------------------------
__global__ void delta_small(const float* __restrict__ dbl, int lddbl, int R,
                            const float* __restrict__ dt_w,
                            const float* __restrict__ dt_b,
                            float* __restrict__ delta, int rows, int Dn) {
    int i = blockIdx.x * blockDim.x + threadIdx.x;
    if (i >= rows * Dn) return;
    int d = i % Dn;
    int row = i / Dn;
    const float* dr = dbl + (size_t)row * lddbl;
    const float* wr = dt_w + (size_t)d * R;
    float acc = dt_b[d];
    for (int r = 0; r < R; ++r) acc += dr[r] * wr[r];
    delta[i] = (acc > 20.f) ? acc : log1pf(expf(acc));
}

// ---------------------------------------------------------------------------
// Sequential selective scan, one thread per (b, d); 8 states in registers.
// dbl[B,L,lddbl]: [dt_rank | Bs(8) | Cs(8)], bcoff = dt_rank.
// Fuses +xc*D and the silu(z) gate; writes ygate[B,L,D].
// ---------------------------------------------------------------------------
__global__ void mamba_scan(const float* __restrict__ delta,
                           const float* __restrict__ dbl, int lddbl, int bcoff,
                           const float* __restrict__ xconv,
                           const float* __restrict__ xz, int ldx, int zoff,
                           const float* __restrict__ Alog, const float* __restrict__ Dp,
                           float* __restrict__ ygate, int Bn, int Ln, int Dn) {
    int idx = blockIdx.x * blockDim.x + threadIdx.x;
    if (idx >= Bn * Dn) return;
    int b = idx / Dn, d = idx % Dn;

    float Acoef[8];
#pragma unroll
    for (int s = 0; s < 8; ++s) Acoef[s] = -expf(Alog[(size_t)d * 8 + s]);
    float h[8] = {0.f, 0.f, 0.f, 0.f, 0.f, 0.f, 0.f, 0.f};
    float Dd = Dp[d];

    for (int l = 0; l < Ln; ++l) {
        size_t row = (size_t)b * Ln + l;
        float dt = delta[row * Dn + d];
        float xc = xconv[row * Dn + d];
        const float* bc = dbl + row * lddbl + bcoff;   // Bs[0..7], Cs[8..15]
        float y = 0.f;
#pragma unroll
        for (int s = 0; s < 8; ++s) {
            float dA = expf(dt * Acoef[s]);
            h[s] = dA * h[s] + dt * bc[s] * xc;
            y += h[s] * bc[8 + s];
        }
        y += xc * Dd;
        float z = xz[row * ldx + zoff + d];
        ygate[row * Dn + d] = y * (z / (1.f + expf(-z)));
    }
}

// ---------------------------------------------------------------------------
// out[b,c,l] = yo1[(b*L+l)*C + c] + yo2[(b*C+c)*L + l] + x[b,c,l]
// ---------------------------------------------------------------------------
__global__ void final_add(const float* __restrict__ yo1, const float* __restrict__ yo2,
                          const float* __restrict__ x, float* __restrict__ out,
                          int Cn, int Ln, int total) {
    int idx = blockIdx.x * blockDim.x + threadIdx.x;
    if (idx >= total) return;
    int l = idx % Ln;
    int c = (idx / Ln) % Cn;
    int b = idx / (Ln * Cn);
    out[idx] = yo1[((size_t)b * Ln + l) * Cn + c] + yo2[idx] + x[idx];
}

// ---------------------------------------------------------------------------
extern "C" void kernel_launch(void* const* d_in, const int* in_sizes, int n_in,
                              void* d_out, int out_size, void* d_ws, size_t ws_size,
                              hipStream_t stream) {
    (void)in_sizes; (void)n_in; (void)out_size; (void)ws_size;
    const int B = 2, C = 128, L = 4096;

    const float* x      = (const float*)d_in[0];
    const float* ln1_g  = (const float*)d_in[1];
    const float* ln1_b  = (const float*)d_in[2];
    const float* ln2_g  = (const float*)d_in[3];
    const float* ln2_b  = (const float*)d_in[4];
    // m1: d_model=128, d_inner=256, dt_rank=8, seq=4096
    const float* m1_in_w   = (const float*)d_in[5];
    const float* m1_conv_w = (const float*)d_in[6];
    const float* m1_conv_b = (const float*)d_in[7];
    const float* m1_xp_w   = (const float*)d_in[8];
    const float* m1_dt_w   = (const float*)d_in[9];
    const float* m1_dt_b   = (const float*)d_in[10];
    const float* m1_Alog   = (const float*)d_in[11];
    const float* m1_D      = (const float*)d_in[12];
    const float* m1_out_w  = (const float*)d_in[13];
    // m2: d_model=4096, d_inner=8192, dt_rank=256, seq=128
    const float* m2_in_w   = (const float*)d_in[14];
    const float* m2_conv_w = (const float*)d_in[15];
    const float* m2_conv_b = (const float*)d_in[16];
    const float* m2_xp_w   = (const float*)d_in[17];
    const float* m2_dt_w   = (const float*)d_in[18];
    const float* m2_dt_b   = (const float*)d_in[19];
    const float* m2_Alog   = (const float*)d_in[20];
    const float* m2_D      = (const float*)d_in[21];
    const float* m2_out_w  = (const float*)d_in[22];
    float* out = (float*)d_out;

    // Workspace layout (floats). Intermediates shared between the two phases.
    float* f = (float*)d_ws;
    size_t o = 0;
    float* YO1   = f + o; o += (size_t)B * L * C;        // m1 result [B,L,C]
    float* YO2   = f + o; o += (size_t)B * C * L;        // m2 result [B,C,L]
    float* NORM  = f + o; o += (size_t)B * L * C;        // max(B*L*128, B*C*4096)
    float* XZ    = f + o; o += (size_t)B * L * 512;      // max(8192*512, 256*16384)
    float* XCV   = f + o; o += (size_t)B * L * 256;      // max(8192*256, 256*8192)
    float* DBL   = f + o; o += (size_t)B * L * 24;       // max(8192*24, 256*272)
    float* DELTA = f + o; o += (size_t)B * L * 256;
    float* YG    = f + o; o += (size_t)B * L * 256;

    auto gemm = [&](const float* A, int lda, const float* W, int ldw,
                    float* Cc, int ldc, int M, int N, int K) {
        int tilesN  = (N + 15) / 16;
        int tilesM4 = (M + 63) / 64;
        int total   = tilesN * tilesM4;
        int blocks  = (total + 7) / 8;          // 8 waves (256 threads) per block
        wmma_gemm_atb<<<blocks, 256, 0, stream>>>(A, lda, W, ldw, Cc, ldc,
                                                  M, N, K, tilesN, tilesM4);
    };
    auto cdiv = [](int a, int b) { return (a + b - 1) / b; };

    // ======================= Mamba 1 (channel mixer) =======================
    const int D1 = 256, R1 = 8, L1 = L, M1rows = B * L1;
    ln_over_c<<<B * L, 128, 0, stream>>>(x, ln1_g, ln1_b, NORM, C, L);
    gemm(NORM, 128, m1_in_w, 128, XZ, 512, M1rows, 512, 128);               // xz
    conv_silu<<<cdiv(B * L1 * D1, 256), 256, 0, stream>>>(XZ, 512, m1_conv_w,
                                                          m1_conv_b, XCV, B, L1, D1);
    gemm(XCV, D1, m1_xp_w, D1, DBL, R1 + 16, M1rows, R1 + 16, D1);          // dbl
    delta_small<<<cdiv(M1rows * D1, 256), 256, 0, stream>>>(DBL, R1 + 16, R1,
                                                            m1_dt_w, m1_dt_b,
                                                            DELTA, M1rows, D1);
    mamba_scan<<<cdiv(B * D1, 128), 128, 0, stream>>>(DELTA, DBL, R1 + 16, R1,
                                                      XCV, XZ, 512, D1,
                                                      m1_Alog, m1_D, YG, B, L1, D1);
    gemm(YG, D1, m1_out_w, D1, YO1, C, M1rows, C, D1);                      // out proj

    // ======================= Mamba 2 (spatial mixer) =======================
    const int D2 = 8192, R2 = 256, L2 = 128, M2rows = B * L2;
    ln_over_l<<<B * C, 256, 0, stream>>>(x, ln2_g, ln2_b, NORM, L);
    gemm(NORM, 4096, m2_in_w, 4096, XZ, 16384, M2rows, 16384, 4096);        // big GEMM
    conv_silu<<<cdiv(B * L2 * D2, 256), 256, 0, stream>>>(XZ, 16384, m2_conv_w,
                                                          m2_conv_b, XCV, B, L2, D2);
    gemm(XCV, D2, m2_xp_w, D2, DBL, R2 + 16, M2rows, R2 + 16, D2);          // dbl
    gemm(DBL, R2 + 16, m2_dt_w, R2, DELTA, D2, M2rows, D2, R2);             // dt@dt_w^T
    bias_softplus<<<cdiv(M2rows * D2, 256), 256, 0, stream>>>(DELTA, m2_dt_b,
                                                              M2rows * D2, D2);
    mamba_scan<<<cdiv(B * D2, 128), 128, 0, stream>>>(DELTA, DBL, R2 + 16, R2,
                                                      XCV, XZ, 16384, D2,
                                                      m2_Alog, m2_D, YG, B, L2, D2);
    gemm(YG, D2, m2_out_w, D2, YO2, 4096, M2rows, 4096, D2);                // big GEMM

    // ============================ Final add ================================
    final_add<<<cdiv(B * C * L, 256), 256, 0, stream>>>(YO1, YO2, x, out,
                                                        C, L, B * C * L);
}
